// TransformerBlock_86268713107537
// MI455X (gfx1250) — matrix operand order repaired and checked by reading, not compiled, not run on hardware
//
#include <hip/hip_runtime.h>

// PointTransformerConv block for MI455X (gfx1250, wave32, WMMA).
// All WMMA operands are staged in LDS in *fragment-ready* layout
// ([tile][lane][16 halves]) so each fragment load is one aligned 32B vector
// load (2x ds_load_b128) instead of 16 scalar ds_load_u16 + packing moves.

typedef __attribute__((ext_vector_type(16))) _Float16 v16h;
typedef __attribute__((ext_vector_type(8)))  float    v8f;

__device__ __forceinline__ v8f wmma_f16(v16h a, v16h b, v8f c) {
  return __builtin_amdgcn_wmma_f32_16x16x32_f16(false, a, false, b, (short)0, c, false, false);
}

// Load one staged fragment: 32 contiguous bytes per lane.
__device__ __forceinline__ v16h frag(const _Float16* buf, int tile, int lane) {
  return *reinterpret_cast<const v16h*>(buf + (((tile << 5) + lane) << 4));
}

// ---- A-fragment index mapping (ISA 7.12.2, 16-bit A 16x32) ----
// lane: m = lane&15, half = lane>>4 ; element e -> k = (e>>3)*16 + half*8 + (e&7)
template <int KS>  // KS = Kdim/32 tiles per row-block
__device__ __forceinline__ int a_pos(int r, int c) {
  const int mt = r >> 4, m = r & 15;
  const int ks = c >> 5, k = c & 31;
  const int half = (k >> 3) & 1;
  const int e    = ((k >> 4) << 3) | (k & 7);
  const int lane = (half << 4) | m;
  return (((mt * KS + ks) << 5) + lane) * 16 + e;
}
template <int KS>
__device__ __forceinline__ void a_inv(int idx, int& r, int& c) {
  const int e    = idx & 15;
  const int lane = (idx >> 4) & 31;
  const int t    = idx >> 9;
  const int mt = t / KS, ks = t % KS;      // KS is a power of two
  const int m = lane & 15, half = lane >> 4;
  const int k = ((e >> 3) << 4) | (half << 3) | (e & 7);
  r = (mt << 4) + m;
  c = (ks << 5) + k;
}

// ---- B-fragment staging (ISA B 32x16): (lane,e) -> k = kt*32+(e>>3)*16+(lane>>4)*8+(e&7),
//      n = nt*16+(lane&15). Tile order: t = kt*NT + nt. W is row-major KdimxNdim f32.
template <int Kdim, int Ndim, int NTHREADS>
__device__ __forceinline__ void stage_w(const float* __restrict__ W,
                                        _Float16* buf, int tid) {
  constexpr int NT = Ndim / 16;
  for (int idx = tid; idx < Kdim * Ndim; idx += NTHREADS) {
    const int e    = idx & 15;
    const int lane = (idx >> 4) & 31;
    const int t    = idx >> 9;
    const int kt = t / NT, nt = t % NT;    // NT is a power of two
    const int k = (kt << 5) + ((e >> 3) << 4) + ((lane >> 4) << 3) + (e & 7);
    const int n = (nt << 4) + (lane & 15);
    buf[idx] = (_Float16)W[k * Ndim + n];
  }
}

#define P_TOT 16384
#define KP1   17
#define E_TOT (P_TOT * KP1)   // 278528, divisible by 64

// ---------------------------------------------------------------------------
// Kernel 1: h = relu(x@Wi+bi); a_src/a_dst/v = h @ {Wsrc,Wdst,Wlin}
// block = 128 thr (4 waves) -> 64 points; grid = P/64.
// ---------------------------------------------------------------------------
__global__ __launch_bounds__(128) void feat_kernel(
    const float* __restrict__ x,
    const float* __restrict__ Wi,   const float* __restrict__ bi,
    const float* __restrict__ Wsrc, const float* __restrict__ Wdst,
    const float* __restrict__ Wlin,
    _Float16* __restrict__ a_src, float* __restrict__ a_dst,
    _Float16* __restrict__ v_out)
{
  __shared__ __align__(32) _Float16 xs[64 * 64];   // A staged (MT=4, KS=2)
  __shared__ __align__(32) _Float16 ws[64 * 128];  // B staged (weights)
  __shared__ __align__(32) _Float16 hs[64 * 64];   // h staged (MT=4, KS=2)
  const int tid = threadIdx.x, lane = tid & 31, wave = tid >> 5;
  const int rows0 = blockIdx.x * 64;

  for (int idx = tid; idx < 64 * 64; idx += 128) {  // x -> A-staged
    int r, c; a_inv<2>(idx, r, c);
    xs[idx] = (_Float16)x[(size_t)(rows0 + r) * 64 + c];
  }
  stage_w<64, 64, 128>(Wi, ws, tid);
  __syncthreads();

  // h = relu(x @ Wi + bi): wave = M-tile, 4 N-tiles, K=64 (2 steps)
#pragma unroll
  for (int nt = 0; nt < 4; ++nt) {
    v8f acc = {};
#pragma unroll
    for (int ks = 0; ks < 2; ++ks)
      acc = wmma_f16(frag(xs, wave * 2 + ks, lane),
                     frag(ws, ks * 4 + nt, lane), acc);
    const int col = nt * 16 + (lane & 15);
    const float bv = bi[col];
#pragma unroll
    for (int rr = 0; rr < 8; ++rr) {
      const int r = wave * 16 + rr + (lane >> 4) * 8;
      const float hv = acc[rr] + bv;
      hs[a_pos<2>(r, col)] = (_Float16)(hv > 0.f ? hv : 0.f);
    }
  }
  __syncthreads();

  const float* Ws[3] = {Wsrc, Wdst, Wlin};
  for (int p = 0; p < 3; ++p) {
    stage_w<64, 128, 128>(Ws[p], ws, tid);
    __syncthreads();
#pragma unroll
    for (int nt = 0; nt < 8; ++nt) {
      v8f acc = {};
#pragma unroll
      for (int ks = 0; ks < 2; ++ks)
        acc = wmma_f16(frag(hs, wave * 2 + ks, lane),
                       frag(ws, ks * 8 + nt, lane), acc);
      const int col = nt * 16 + (lane & 15);
#pragma unroll
      for (int rr = 0; rr < 8; ++rr) {
        const size_t row = (size_t)(rows0 + wave * 16 + rr + (lane >> 4) * 8);
        if (p == 0)      a_src[row * 128 + col] = (_Float16)acc[rr];
        else if (p == 1) a_dst[row * 128 + col] = acc[rr];
        else             v_out[row * 128 + col] = (_Float16)acc[rr];
      }
    }
    __syncthreads();
  }
}

// ---------------------------------------------------------------------------
// Kernel 2: per-cloud brute-force kNN (k=16, self excluded) + self loop slot 0.
// ---------------------------------------------------------------------------
__global__ __launch_bounds__(256) void knn_kernel(
    const float* __restrict__ pos, int* __restrict__ nbr)
{
  __shared__ float cp[256 * 3];
  const int tid = threadIdx.x;
  const int cloud = blockIdx.x >> 3;
  const int q  = ((blockIdx.x & 7) << 8) + tid;
  const int gq = cloud * 2048 + q;
  const float qx = pos[gq * 3 + 0], qy = pos[gq * 3 + 1], qz = pos[gq * 3 + 2];

  float dist[16]; int didx[16];
#pragma unroll
  for (int s = 0; s < 16; ++s) { dist[s] = 3.0e38f; didx[s] = q; }

  for (int t0 = 0; t0 < 2048; t0 += 256) {
    __syncthreads();
    const int gc = cloud * 2048 + t0 + tid;
    cp[tid * 3 + 0] = pos[gc * 3 + 0];
    cp[tid * 3 + 1] = pos[gc * 3 + 1];
    cp[tid * 3 + 2] = pos[gc * 3 + 2];
    __syncthreads();
    for (int ct = 0; ct < 256; ++ct) {
      const int j = t0 + ct;
      if (j == q) continue;
      const float dx = qx - cp[ct * 3 + 0];
      const float dy = qy - cp[ct * 3 + 1];
      const float dz = qz - cp[ct * 3 + 2];
      const float d = dx * dx + dy * dy + dz * dz;
      if (d < dist[15]) {
        dist[15] = d; didx[15] = j;
#pragma unroll
        for (int s = 15; s > 0; --s) {
          if (dist[s] < dist[s - 1]) {
            const float td = dist[s]; dist[s] = dist[s - 1]; dist[s - 1] = td;
            const int   ti = didx[s]; didx[s] = didx[s - 1]; didx[s - 1] = ti;
          }
        }
      }
    }
  }
  nbr[gq * KP1] = gq;
#pragma unroll
  for (int s = 0; s < 16; ++s) nbr[gq * KP1 + 1 + s] = cloud * 2048 + didx[s];
}

// ---------------------------------------------------------------------------
// Kernel 3: fused per-edge chain (64 edge rows per block, 4 waves).
//   t1 = relu(rel@Wp1+bp1)              (scalar K=3 -> A-staged)
//   delta = relu(t1@Wp2+bp2)            (WMMA -> global f16 + A-staged)
//   alpha = a_dst_i - a_src_j + delta   (in-place on staged buffer)
//   t2 = relu(alpha@Wa1+ba1)            (WMMA K=128 -> A-staged)
//   attn = relu(t2@Wa2+ba2)             (WMMA -> global f16)
// ---------------------------------------------------------------------------
__global__ __launch_bounds__(128) void edge_kernel(
    const float* __restrict__ pos, const int* __restrict__ nbr,
    const _Float16* __restrict__ a_src, const float* __restrict__ a_dst,
    const float* __restrict__ Wp1, const float* __restrict__ bp1,
    const float* __restrict__ Wp2, const float* __restrict__ bp2,
    const float* __restrict__ Wa1, const float* __restrict__ ba1,
    const float* __restrict__ Wa2, const float* __restrict__ ba2,
    _Float16* __restrict__ delta_g, _Float16* __restrict__ attn_g)
{
  __shared__ float relf[64 * 4];
  __shared__ __align__(32) _Float16 t1s[64 * 64];   // t1 staged, later t2 staged
  __shared__ __align__(32) _Float16 ds_[64 * 128];  // delta/alpha staged (KS=4)
  __shared__ __align__(32) _Float16 ws[8192];       // staged weights (reused)
  const int tid = threadIdx.x, lane = tid & 31, wave = tid >> 5;
  const int rows0 = blockIdx.x * 64;

  if (tid < 64) {
    const int row = rows0 + tid;
    const int i = row / KP1;
    const int j = nbr[row];
    relf[tid * 4 + 0] = pos[i * 3 + 0] - pos[j * 3 + 0];
    relf[tid * 4 + 1] = pos[i * 3 + 1] - pos[j * 3 + 1];
    relf[tid * 4 + 2] = pos[i * 3 + 2] - pos[j * 3 + 2];
  }
  stage_w<64, 128, 128>(Wp2, ws, tid);
  __syncthreads();

  for (int idx = tid; idx < 64 * 64; idx += 128) {   // t1 (K=3, scalar)
    int r, c; a_inv<2>(idx, r, c);
    float t = bp1[c] + relf[r * 4 + 0] * Wp1[c]
                     + relf[r * 4 + 1] * Wp1[64 + c]
                     + relf[r * 4 + 2] * Wp1[128 + c];
    t1s[idx] = (_Float16)(t > 0.f ? t : 0.f);
  }
  __syncthreads();

#pragma unroll
  for (int nt = 0; nt < 8; ++nt) {                   // delta = relu(t1@Wp2+bp2)
    v8f acc = {};
#pragma unroll
    for (int ks = 0; ks < 2; ++ks)
      acc = wmma_f16(frag(t1s, wave * 2 + ks, lane),
                     frag(ws, ks * 8 + nt, lane), acc);
    const int col = nt * 16 + (lane & 15);
    const float bv = bp2[col];
#pragma unroll
    for (int rr = 0; rr < 8; ++rr) {
      const int r = wave * 16 + rr + (lane >> 4) * 8;
      float dv = acc[rr] + bv; dv = dv > 0.f ? dv : 0.f;
      ds_[a_pos<4>(r, col)] = (_Float16)dv;
      delta_g[(size_t)(rows0 + r) * 128 + col] = (_Float16)dv;
    }
  }
  __syncthreads();

  stage_w<128, 64, 128>(Wa1, ws, tid);               // Wa1 staged (KT=4, NT=4)
  for (int idx = tid; idx < 64 * 128; idx += 128) {  // alpha in place
    int r, c; a_inv<4>(idx, r, c);
    const int row = rows0 + r;
    const int i = row / KP1;
    const int j = nbr[row];
    const float av = a_dst[(size_t)i * 128 + c]
                   - (float)a_src[(size_t)j * 128 + c]
                   + (float)ds_[idx];
    ds_[idx] = (_Float16)av;
  }
  __syncthreads();

#pragma unroll
  for (int nt = 0; nt < 4; ++nt) {                   // t2 = relu(alpha@Wa1+ba1)
    v8f acc = {};
#pragma unroll
    for (int ks = 0; ks < 4; ++ks)
      acc = wmma_f16(frag(ds_, wave * 4 + ks, lane),
                     frag(ws, ks * 4 + nt, lane), acc);
    const int col = nt * 16 + (lane & 15);
    const float bv = ba1[col];
#pragma unroll
    for (int rr = 0; rr < 8; ++rr) {
      const int r = wave * 16 + rr + (lane >> 4) * 8;
      const float tv = acc[rr] + bv;
      t1s[a_pos<2>(r, col)] = (_Float16)(tv > 0.f ? tv : 0.f);
    }
  }
  __syncthreads();

  stage_w<64, 128, 128>(Wa2, ws, tid);               // Wa2 staged (KT=2, NT=8)
  __syncthreads();

#pragma unroll
  for (int nt = 0; nt < 8; ++nt) {                   // attn = relu(t2@Wa2+ba2)
    v8f acc = {};
#pragma unroll
    for (int ks = 0; ks < 2; ++ks)
      acc = wmma_f16(frag(t1s, wave * 2 + ks, lane),
                     frag(ws, ks * 8 + nt, lane), acc);
    const int col = nt * 16 + (lane & 15);
    const float bv = ba2[col];
#pragma unroll
    for (int rr = 0; rr < 8; ++rr) {
      const int r = wave * 16 + rr + (lane >> 4) * 8;
      const float av = acc[rr] + bv;
      attn_g[(size_t)(rows0 + r) * 128 + col] = (_Float16)(av > 0.f ? av : 0.f);
    }
  }
}

// ---------------------------------------------------------------------------
// Kernel 4: per-(point, channel) softmax over 17 edges + mean of w*(v_j+delta)
// ---------------------------------------------------------------------------
__global__ __launch_bounds__(256) void softmax_msg_kernel(
    const _Float16* __restrict__ attn_g, const _Float16* __restrict__ delta_g,
    const _Float16* __restrict__ v_g, const int* __restrict__ nbr,
    _Float16* __restrict__ out_pre)
{
  const int gid = blockIdx.x * 256 + threadIdx.x;  // < P*128
  const int i = gid >> 7, c = gid & 127;
  const int e0 = i * KP1;
  float a[KP1];
  float mx = -3.0e38f;
#pragma unroll
  for (int k = 0; k < KP1; ++k) {
    a[k] = (float)attn_g[(size_t)(e0 + k) * 128 + c];
    mx = a[k] > mx ? a[k] : mx;
  }
  float s = 0.f;
#pragma unroll
  for (int k = 0; k < KP1; ++k) { a[k] = __expf(a[k] - mx); s += a[k]; }
  float acc = 0.f;
#pragma unroll
  for (int k = 0; k < KP1; ++k) {
    const int j = nbr[e0 + k];
    acc += a[k] * ((float)v_g[(size_t)j * 128 + c]
                 + (float)delta_g[(size_t)(e0 + k) * 128 + c]);
  }
  out_pre[gid] = (_Float16)(acc / (s * (float)KP1));
}

// ---------------------------------------------------------------------------
// Kernel 5: out = relu(out_pre @ Wo + bo)  (K=128 WMMA) -> f32 d_out
// ---------------------------------------------------------------------------
__global__ __launch_bounds__(128) void linout_kernel(
    const _Float16* __restrict__ out_pre,
    const float* __restrict__ Wo, const float* __restrict__ bo,
    float* __restrict__ out)
{
  __shared__ __align__(32) _Float16 as_[64 * 128];   // A staged (MT=4, KS=4)
  __shared__ __align__(32) _Float16 ws[128 * 128];   // Wo staged (KT=4, NT=8)
  const int tid = threadIdx.x, lane = tid & 31, wave = tid >> 5;
  const int rows0 = blockIdx.x * 64;

  for (int idx = tid; idx < 64 * 128; idx += 128) {
    int r, c; a_inv<4>(idx, r, c);
    as_[idx] = out_pre[(size_t)(rows0 + r) * 128 + c];
  }
  stage_w<128, 128, 128>(Wo, ws, tid);
  __syncthreads();

#pragma unroll
  for (int nt = 0; nt < 8; ++nt) {
    v8f acc = {};
#pragma unroll
    for (int ks = 0; ks < 4; ++ks)
      acc = wmma_f16(frag(as_, wave * 4 + ks, lane),
                     frag(ws, ks * 8 + nt, lane), acc);
    const int col = nt * 16 + (lane & 15);
    const float bv = bo[col];
#pragma unroll
    for (int rr = 0; rr < 8; ++rr) {
      const int r = wave * 16 + rr + (lane >> 4) * 8;
      const float ov = acc[rr] + bv;
      out[(size_t)(rows0 + r) * 128 + col] = ov > 0.f ? ov : 0.f;
    }
  }
}

// ---------------------------------------------------------------------------
extern "C" void kernel_launch(void* const* d_in, const int* in_sizes, int n_in,
                              void* d_out, int out_size, void* d_ws, size_t ws_size,
                              hipStream_t stream)
{
  (void)in_sizes; (void)n_in; (void)out_size; (void)ws_size;
  const float* x    = (const float*)d_in[0];
  const float* pos  = (const float*)d_in[1];
  // d_in[2] = batch: unused (clouds are contiguous blocks of N=2048)
  const float* Wi   = (const float*)d_in[3];
  const float* bi   = (const float*)d_in[4];
  const float* Wsrc = (const float*)d_in[5];
  const float* Wdst = (const float*)d_in[6];
  const float* Wlin = (const float*)d_in[7];
  const float* Wp1  = (const float*)d_in[8];
  const float* bp1  = (const float*)d_in[9];
  const float* Wp2  = (const float*)d_in[10];
  const float* bp2  = (const float*)d_in[11];
  const float* Wa1  = (const float*)d_in[12];
  const float* ba1  = (const float*)d_in[13];
  const float* Wa2  = (const float*)d_in[14];
  const float* ba2  = (const float*)d_in[15];
  const float* Wo   = (const float*)d_in[16];
  const float* bo   = (const float*)d_in[17];

  char* ws = (char*)d_ws;
  size_t off = 0;
  auto take = [&](size_t bytes) -> char* {
    char* p = ws + off;
    off += (bytes + 255) & ~(size_t)255;
    return p;
  };
  _Float16* a_src = (_Float16*)take((size_t)P_TOT * 128 * 2);
  float*    a_dst = (float*)   take((size_t)P_TOT * 128 * 4);
  _Float16* v_g   = (_Float16*)take((size_t)P_TOT * 128 * 2);
  int*      nbr   = (int*)     take((size_t)E_TOT * 4);
  _Float16* delta = (_Float16*)take((size_t)E_TOT * 128 * 2);
  _Float16* attn  = (_Float16*)take((size_t)E_TOT * 128 * 2);
  _Float16* outp  = (_Float16*)take((size_t)P_TOT * 128 * 2);

  feat_kernel<<<P_TOT / 64, 128, 0, stream>>>(x, Wi, bi, Wsrc, Wdst, Wlin,
                                              a_src, a_dst, v_g);
  knn_kernel<<<64, 256, 0, stream>>>(pos, nbr);
  edge_kernel<<<E_TOT / 64, 128, 0, stream>>>(pos, nbr, a_src, a_dst,
                                              Wp1, bp1, Wp2, bp2,
                                              Wa1, ba1, Wa2, ba2,
                                              delta, attn);
  softmax_msg_kernel<<<(P_TOT * 128) / 256, 256, 0, stream>>>(attn, delta, v_g,
                                                              nbr, outp);
  linout_kernel<<<P_TOT / 64, 128, 0, stream>>>(outp, Wo, bo, (float*)d_out);
}
